// EmbedLayerUpdate_43774306680842
// MI455X (gfx1250) — matrix operand order repaired and checked by reading, not compiled, not run on hardware
//
#include <hip/hip_runtime.h>

// ---------------------------------------------------------------------------
// Problem constants (match reference): B=16, D0=2048, D1=D2=4096, P=64
// ---------------------------------------------------------------------------
#define BATCH 16
#define DIM   4096
#define PDIM  64

typedef __attribute__((ext_vector_type(16))) __bf16 v16bf;
typedef __attribute__((ext_vector_type(4)))  __bf16 v4bf;
typedef __attribute__((ext_vector_type(8)))  float  v8f;

static __device__ inline v8f v8f_zero() {
    v8f z;
#pragma unroll
    for (int i = 0; i < 8; ++i) z[i] = 0.0f;
    return z;
}

// A-fragment layout within one 32-wide K chunk:
//   fragment element i of lane-group g holds k = 16*g + i
//   inverse: position p = k (identity, matches the [n][k] B fragments)
// Any 4-aligned run of k maps to 4 consecutive p -> v4bf stores stay legal.
static __device__ inline int swz32(int kl) {
    return kl;
}

static __device__ inline v4bf cvt4(float4 v) {
    v4bf o;
    o[0] = (__bf16)v.x; o[1] = (__bf16)v.y;
    o[2] = (__bf16)v.z; o[3] = (__bf16)v.w;
    return o;
}

// ---------------------------------------------------------------------------
// GEMM:  Y[b] = W[M x K] @ X[b][K x 64] + bias   (fp32 in, bf16 WMMA, fp32 out)
// 256 threads = 8 waves; M-tile 128 (16 rows/wave), N = 64, K-step 32.
// LDS tiles are stored in WMMA fragment order: each lane's fragment is one
// 32-byte contiguous v16bf load (2 x ds_load_b128), no element gathers.
// ---------------------------------------------------------------------------
__global__ __launch_bounds__(256)
void gemm_bf16_wmma(const float* __restrict__ W,    // [M x K]
                    const float* __restrict__ X,    // [B x K x 64]
                    const float* __restrict__ bias, // [M]
                    float* __restrict__ Y,          // [B x M x 64]
                    int M, int K)
{
    __shared__ __align__(32) __bf16 As[128][32];  // [m][swz32(k)]
    __shared__ __align__(32) __bf16 Bt[64][32];   // [n][k]  (transposed X tile)

    const int b    = blockIdx.y;
    const int m0   = blockIdx.x * 128;
    const int tid  = threadIdx.x;
    const int lane = tid & 31;
    const int wave = tid >> 5;
    const int g    = lane >> 4;
    const int hl   = lane & 15;

    const float* Xb = X + (size_t)b * K * 64;

    v8f acc[4];
#pragma unroll
    for (int nt = 0; nt < 4; ++nt) acc[nt] = v8f_zero();

    for (int k0 = 0; k0 < K; k0 += 32) {
        // stage W tile (128x32): float4 global loads, swizzled v4bf LDS stores
#pragma unroll
        for (int it = 0; it < 4; ++it) {
            int flat = (tid + it * 256) * 4;          // 0..4095 step 4
            int r = flat >> 5, c = flat & 31;
            float4 v = *(const float4*)&W[(size_t)(m0 + r) * K + (k0 + c)];
            *(v4bf*)&As[r][swz32(c)] = cvt4(v);
        }
        // stage X tile (32x64): float4 loads, transpose-scatter bf16 stores
#pragma unroll
        for (int it = 0; it < 2; ++it) {
            int flat = (tid + it * 256) * 4;          // 0..2047 step 4
            int r = flat >> 6, c = flat & 63;
            float4 v = *(const float4*)&Xb[(size_t)(k0 + r) * 64 + c];
            Bt[c + 0][r] = (__bf16)v.x;
            Bt[c + 1][r] = (__bf16)v.y;
            Bt[c + 2][r] = (__bf16)v.z;
            Bt[c + 3][r] = (__bf16)v.w;
        }
        __syncthreads();

        // prefetch next W tile (global_prefetch_b8)
        if (k0 + 32 < K) {
            int idx = tid << 4;
            int r = idx >> 5, c = idx & 31;
            __builtin_prefetch(&W[(size_t)(m0 + r) * K + (k0 + 32 + c)], 0, 3);
        }

        const v16bf a = *(const v16bf*)&As[wave * 16 + hl][g * 16];
#pragma unroll
        for (int nt = 0; nt < 4; ++nt) {
            const v16bf bf = *(const v16bf*)&Bt[nt * 16 + hl][g * 16];
            acc[nt] = __builtin_amdgcn_wmma_f32_16x16x32_bf16(
                false, a, false, bf, (short)0, acc[nt], false, false);
        }
        __syncthreads();
    }

    // epilogue: C/D layout — lane holds col n = nt*16+hl, VGPR i -> m = i+8g
    const int mbase = m0 + wave * 16;
#pragma unroll
    for (int nt = 0; nt < 4; ++nt) {
#pragma unroll
        for (int i = 0; i < 8; ++i) {
            int m = mbase + i + 8 * g;
            int n = nt * 16 + hl;
            Y[((size_t)b * M + m) * 64 + n] = acc[nt][i] + bias[m];
        }
    }
}

// ---------------------------------------------------------------------------
// Fused relu-block (per row of [N=B*D, 64]):
//   info = [ur, lr, diff, ps, ss]
//   h    = relu(info @ fc1^T + b1) @ fc11^T + b11      (scalar K=5, WMMA K=64)
//   e    = [pre, h]
//   out  = (relu(e @ fc2^T + b2) @ fc21^T + b21)*mask  (WMMA K=128, K=64)
// 128 threads = 4 waves, 16 rows/wave. fcXX_W row-major [o][k] is already the
// B-transposed layout -> direct bf16 copy; fragments are contiguous v16bf.
// Activation staging (tS/eS) uses the chunk-swizzled A layout.
// ---------------------------------------------------------------------------
__global__ __launch_bounds__(128)
void relu_block_wmma(const float* __restrict__ pre,   // [N x 64]
                     const float* __restrict__ lb,    // [N]
                     const float* __restrict__ ub,    // [N]
                     const float* __restrict__ ps,    // [N]
                     const float* __restrict__ ss,    // [N]
                     const float* __restrict__ mask,  // [N]
                     const float* __restrict__ fc1W,  // [64 x 5]
                     const float* __restrict__ fc1b,  // [64]
                     const float* __restrict__ fc11W, // [64 x 64]
                     const float* __restrict__ fc11b, // [64]
                     const float* __restrict__ fc2W,  // [64 x 128]
                     const float* __restrict__ fc2b,  // [64]
                     const float* __restrict__ fc21W, // [64 x 64]
                     const float* __restrict__ fc21b, // [64]
                     float* __restrict__ out)         // [N x 64]
{
    __shared__ float sfc1W[64 * 5];
    __shared__ float sb1[64], sb11[64], sb2[64], sb21[64];
    __shared__ __align__(32) __bf16 sW11[64 * 64];    // [o][k] (= B^T layout)
    __shared__ __align__(32) __bf16 sW2[64 * 128];    // [o][k]
    __shared__ __align__(32) __bf16 sW21[64 * 64];    // [o][k]
    __shared__ float sinfo[4][16][5];
    __shared__ __align__(32) __bf16 eS[4][16][128];   // chunk-swizzled A staging
    __shared__ __align__(32) __bf16 tS[4][16][64];    // chunk-swizzled A staging

    const int tid  = threadIdx.x;
    const int lane = tid & 31;
    const int wave = tid >> 5;
    const int g    = lane >> 4;
    const int hl   = lane & 15;

    // ---- weight staging: float4 loads, bf16x4 stores, no transpose needed ----
    for (int i = tid; i < 64 * 5; i += 128) sfc1W[i] = fc1W[i];
    if (tid < 64) {
        sb1[tid]  = fc1b[tid];
        sb11[tid] = fc11b[tid];
        sb2[tid]  = fc2b[tid];
        sb21[tid] = fc21b[tid];
    }
#pragma unroll
    for (int it = 0; it < 8; ++it) {
        int flat = (tid + it * 128) * 4;              // 0..4095 step 4
        *(v4bf*)&sW11[flat] = cvt4(*(const float4*)&fc11W[flat]);
        *(v4bf*)&sW21[flat] = cvt4(*(const float4*)&fc21W[flat]);
    }
#pragma unroll
    for (int it = 0; it < 16; ++it) {
        int flat = (tid + it * 128) * 4;              // 0..8191 step 4
        *(v4bf*)&sW2[flat] = cvt4(*(const float4*)&fc2W[flat]);
    }
    __syncthreads();

    const int rowBase = blockIdx.x * 64 + wave * 16;

    // ---- info features (lanes 0..15, one row each) ----
    if (lane < 16) {
        int r = rowBase + lane;
        float l = lb[r], u = ub[r];
        float lt = fminf(l, 0.0f), ut = fmaxf(u, 0.0f);
        float diff = ut - lt;
        float urr  = ut / (diff + 1e-8f);
        sinfo[wave][lane][0] = urr;
        sinfo[wave][lane][1] = 1.0f - urr;
        sinfo[wave][lane][2] = diff;
        sinfo[wave][lane][3] = ps[r];
        sinfo[wave][lane][4] = ss[r];
    }

    // ---- h1 = relu(info @ fc1^T + b1): 16x64 scalar (K=5), swizzled stage ----
#pragma unroll 4
    for (int j = 0; j < 32; ++j) {
        int flat = lane + 32 * j;                     // 0..1023
        int r = flat >> 6, o = flat & 63;
        float a = sb1[o];
#pragma unroll
        for (int k = 0; k < 5; ++k) a += sinfo[wave][r][k] * sfc1W[o * 5 + k];
        tS[wave][r][((o >> 5) << 5) + swz32(o & 31)] = (__bf16)fmaxf(a, 0.0f);
    }
    // ---- stage pre into e[:, 0:64] (float4 loads, swizzled v4bf stores) ----
    {
        const float* prow = pre + (size_t)rowBase * 64;
#pragma unroll
        for (int j = 0; j < 8; ++j) {
            int flat = (lane + 32 * j) * 4;           // 0..1023 step 4
            int r = flat >> 6, c = flat & 63;
            float4 v = *(const float4*)&prow[r * 64 + c];
            *(v4bf*)&eS[wave][r][((c >> 5) << 5) + swz32(c & 31)] = cvt4(v);
        }
    }

    v8f acc[4];

    // ---- WMMA 1: h = h1 @ fc11^T + b11 (K=64) -> e[:, 64:128] ----
#pragma unroll
    for (int nt = 0; nt < 4; ++nt) acc[nt] = v8f_zero();
#pragma unroll
    for (int ks = 0; ks < 64; ks += 32) {
        const v16bf a = *(const v16bf*)&tS[wave][hl][ks + g * 16];
#pragma unroll
        for (int nt = 0; nt < 4; ++nt) {
            int col = nt * 16 + hl;
            const v16bf bf = *(const v16bf*)&sW11[col * 64 + ks + g * 16];
            acc[nt] = __builtin_amdgcn_wmma_f32_16x16x32_bf16(
                false, a, false, bf, (short)0, acc[nt], false, false);
        }
    }
#pragma unroll
    for (int nt = 0; nt < 4; ++nt)
#pragma unroll
        for (int i = 0; i < 8; ++i) {
            int m = i + 8 * g, n = nt * 16 + hl;
            eS[wave][m][64 + ((n >> 5) << 5) + swz32(n & 31)] =
                (__bf16)(acc[nt][i] + sb11[n]);
        }

    // ---- WMMA 2: t = relu(e @ fc2^T + b2) (K=128) ----
#pragma unroll
    for (int nt = 0; nt < 4; ++nt) acc[nt] = v8f_zero();
#pragma unroll
    for (int ks = 0; ks < 128; ks += 32) {
        const v16bf a = *(const v16bf*)&eS[wave][hl][ks + g * 16];
#pragma unroll
        for (int nt = 0; nt < 4; ++nt) {
            int col = nt * 16 + hl;
            const v16bf bf = *(const v16bf*)&sW2[col * 128 + ks + g * 16];
            acc[nt] = __builtin_amdgcn_wmma_f32_16x16x32_bf16(
                false, a, false, bf, (short)0, acc[nt], false, false);
        }
    }
#pragma unroll
    for (int nt = 0; nt < 4; ++nt)
#pragma unroll
        for (int i = 0; i < 8; ++i) {
            int m = i + 8 * g, n = nt * 16 + hl;
            tS[wave][m][((n >> 5) << 5) + swz32(n & 31)] =
                (__bf16)fmaxf(acc[nt][i] + sb2[n], 0.0f);
        }

    // ---- WMMA 3: out = (t @ fc21^T + b21) * mask (K=64) ----
#pragma unroll
    for (int nt = 0; nt < 4; ++nt) acc[nt] = v8f_zero();
#pragma unroll
    for (int ks = 0; ks < 64; ks += 32) {
        const v16bf a = *(const v16bf*)&tS[wave][hl][ks + g * 16];
#pragma unroll
        for (int nt = 0; nt < 4; ++nt) {
            int col = nt * 16 + hl;
            const v16bf bf = *(const v16bf*)&sW21[col * 64 + ks + g * 16];
            acc[nt] = __builtin_amdgcn_wmma_f32_16x16x32_bf16(
                false, a, false, bf, (short)0, acc[nt], false, false);
        }
    }
    float* orow = out + (size_t)rowBase * 64;
#pragma unroll
    for (int nt = 0; nt < 4; ++nt)
#pragma unroll
        for (int i = 0; i < 8; ++i) {
            int m = i + 8 * g, n = nt * 16 + hl;
            orow[m * 64 + n] = (acc[nt][i] + sb21[n]) * mask[rowBase + m];
        }
}

// ---------------------------------------------------------------------------
// Launch: GEMM1 -> relu1 (mu1 straight into d_out, consumed by GEMM2) ->
//         GEMM2 -> relu2 (mu2 into d_out second half).
// Workspace: one [B*D*64] fp32 pre-activation buffer (16 MiB).
// ---------------------------------------------------------------------------
extern "C" void kernel_launch(void* const* d_in, const int* in_sizes, int n_in,
                              void* d_out, int out_size, void* d_ws, size_t ws_size,
                              hipStream_t stream) {
    (void)in_sizes; (void)n_in; (void)out_size; (void)ws_size;

    const float* mu0   = (const float*)d_in[0];
    const float* lb1   = (const float*)d_in[1];
    const float* ub1   = (const float*)d_in[2];
    const float* lb2   = (const float*)d_in[3];
    const float* ub2   = (const float*)d_in[4];
    const float* ps1   = (const float*)d_in[5];
    const float* ss1   = (const float*)d_in[6];
    const float* ps2   = (const float*)d_in[7];
    const float* ss2   = (const float*)d_in[8];
    const float* mask1 = (const float*)d_in[9];
    const float* mask2 = (const float*)d_in[10];
    const float* W1    = (const float*)d_in[11];
    const float* b1    = (const float*)d_in[12];
    const float* W2    = (const float*)d_in[13];
    const float* b2    = (const float*)d_in[14];
    const float* fc1W  = (const float*)d_in[15];
    const float* fc1b  = (const float*)d_in[16];
    const float* fc11W = (const float*)d_in[17];
    const float* fc11b = (const float*)d_in[18];
    const float* fc2W  = (const float*)d_in[19];
    const float* fc2b  = (const float*)d_in[20];
    const float* fc21W = (const float*)d_in[21];
    const float* fc21b = (const float*)d_in[22];

    float* out = (float*)d_out;
    float* mu1 = out;                               // [B, D, 64]
    float* mu2 = out + (size_t)BATCH * DIM * PDIM;  // [B, D, 64]
    float* pre = (float*)d_ws;                      // [B*D, 64] fp32

    dim3 gGrid(DIM / 128, BATCH);
    const int rGrid = (BATCH * DIM) / 64;

    // Layer 1
    gemm_bf16_wmma<<<gGrid, 256, 0, stream>>>(W1, mu0, b1, pre, DIM, 2048);
    relu_block_wmma<<<rGrid, 128, 0, stream>>>(pre, lb1, ub1, ps1, ss1, mask1,
        fc1W, fc1b, fc11W, fc11b, fc2W, fc2b, fc21W, fc21b, mu1);

    // Layer 2 (reads mu1 fp32 from d_out, converts to bf16 on the fly)
    gemm_bf16_wmma<<<gGrid, 256, 0, stream>>>(W2, mu1, b2, pre, DIM, DIM);
    relu_block_wmma<<<rGrid, 128, 0, stream>>>(pre, lb2, ub2, ps2, ss2, mask2,
        fc1W, fc1b, fc11W, fc11b, fc2W, fc2b, fc21W, fc21b, mu2);
}
